// SparseConvolutionDownsample_25967372272144
// MI455X (gfx1250) — matrix operand order repaired and checked by reading, not compiled, not run on hardware
//
#include <hip/hip_runtime.h>

#define CIN   128
#define COUT  256
#define KOFF  8
#define NIN   400000
#define MOUT  100000
#define MTILE 64
#define NWG   1563   // ceil(100000/64)
#define EPS   1e-4f

typedef __attribute__((ext_vector_type(2)))  float  v2f;
typedef __attribute__((ext_vector_type(4)))  float  v4f;
typedef __attribute__((ext_vector_type(8)))  float  v8f;
typedef __attribute__((ext_vector_type(16))) __bf16 v16bf;

typedef v16bf LfeT[4][4][2][32];   // [chunk][msub][hl][lane], per k

static __device__ __forceinline__ v8f wmma_bf16(v16bf a, v16bf b, v8f c) {
  return __builtin_amdgcn_wmma_f32_16x16x32_bf16(
      false, a, false, b, (short)0, c, false, false);
}

// ---------------------------------------------------------------------------
// Kernel 1: build W as hi/lo bf16 planes in WMMA B-fragment order.
// Wsw[k][chunk][nt][hl][lane] : 16 bf16 (32 B) per lane.
//   n = nt*16 + (lane&15)
//   lane 0-15 holds K = chunk*32 + 0..15 ; lane 16-31 holds K = +16..31
//   element j (VGPR pair order) = K = chunk*32 + (lane>>4)*16 + j
// ---------------------------------------------------------------------------
__global__ __launch_bounds__(256) void build_wsw(const float* __restrict__ W,
                                                 v16bf* __restrict__ Wsw) {
  int idx   = blockIdx.x * 256 + threadIdx.x;  // 8*4*16*2*32 = 32768 entries
  int lane  = idx & 31;
  int hl    = (idx >> 5) & 1;
  int nt    = (idx >> 6) & 15;
  int chunk = (idx >> 10) & 3;
  int k     = idx >> 12;
  int n     = nt * 16 + (lane & 15);
  int kb    = chunk * 32 + (lane >> 4) * 16;
  v16bf v;
#pragma unroll
  for (int j = 0; j < 16; ++j) {
    float  w = W[(k * CIN + kb + j) * COUT + n];
    __bf16 h = (__bf16)w;
    v[j] = hl ? (__bf16)(w - (float)h) : h;
  }
  Wsw[idx] = v;
}

// ---------------------------------------------------------------------------
// Compute body with compile-time M-subtile base (immediate LDS offsets).
// 12 WMMAs per K-chunk, round-robin over the 4 accumulators so each
// dependency chain has distance 4 and hi fragments retire early.
// ---------------------------------------------------------------------------
template <int MS0>
static __device__ __forceinline__ void compute_tile(
    const LfeT* __restrict__ lfe, const v16bf* __restrict__ Wsw,
    int lane, int nt0, v8f& acc00, v8f& acc01, v8f& acc10, v8f& acc11) {
  for (int k = 0; k < KOFF; ++k) {
#pragma unroll
    for (int chunk = 0; chunk < 4; ++chunk) {
      v16bf a0h = lfe[k][chunk][MS0][0][lane];
      v16bf a0l = lfe[k][chunk][MS0][1][lane];
      v16bf a1h = lfe[k][chunk][MS0 + 1][0][lane];
      v16bf a1l = lfe[k][chunk][MS0 + 1][1][lane];
      const v16bf* bp = Wsw + ((((k * 4 + chunk) * 16 + nt0) * 2) * 32 + lane);
      v16bf b0h = bp[0];
      v16bf b0l = bp[32];
      v16bf b1h = bp[64];
      v16bf b1l = bp[96];
      // bf16x3: hi*hi + hi*lo + lo*hi  (lo*lo ~ 2^-16, dropped)
      acc00 = wmma_bf16(a0h, b0h, acc00);
      acc01 = wmma_bf16(a0h, b1h, acc01);
      acc10 = wmma_bf16(a1h, b0h, acc10);
      acc11 = wmma_bf16(a1h, b1h, acc11);
      acc00 = wmma_bf16(a0h, b0l, acc00);
      acc01 = wmma_bf16(a0h, b1l, acc01);
      acc10 = wmma_bf16(a1h, b0l, acc10);
      acc11 = wmma_bf16(a1h, b1l, acc11);
      acc00 = wmma_bf16(a0l, b0h, acc00);
      acc01 = wmma_bf16(a0l, b1h, acc01);
      acc10 = wmma_bf16(a1l, b0h, acc10);
      acc11 = wmma_bf16(a1l, b1h, acc11);
    }
  }
}

// ---------------------------------------------------------------------------
// Kernel 2: gather + GEMM (bf16x3 split precision, f32 accumulate).
// One workgroup = 64 output rows x 256 cols, 512 threads = 16 waves.
// Wave w owns N-tile pair (w&7)*2 and M-subtile pair (w>>3)*2.
// ---------------------------------------------------------------------------
__global__ __launch_bounds__(512) void gemm_kernel(
    const float* __restrict__ feats, const int* __restrict__ gidx,
    const v16bf* __restrict__ Wsw, float* __restrict__ out) {
  __shared__ v16bf lfe[KOFF][4][4][2][32];     // 256 KB
  const int t  = threadIdx.x;
  const int m0 = blockIdx.x * MTILE;

  // ---- stage gathered features: one thread per (row, k) pair ----
  {
    int row  = t & 63;
    int k    = t >> 6;
    int msub = row >> 4;
    int r    = row & 15;
    bool rowvalid = (m0 + row) < MOUT;
    int gi   = rowvalid ? gidx[(m0 + row) * KOFF + k] : NIN;
    const v4f* src = (const v4f*)&feats[(long)gi * CIN];
    bool valid = (gi < NIN);
#pragma unroll
    for (int chunk = 0; chunk < 4; ++chunk) {
      float x[32];
      if (valid) {
#pragma unroll
        for (int i = 0; i < 8; ++i) {
          v4f v = src[chunk * 8 + i];
          x[i * 4 + 0] = v.x; x[i * 4 + 1] = v.y;
          x[i * 4 + 2] = v.z; x[i * 4 + 3] = v.w;
        }
      } else {
#pragma unroll
        for (int c = 0; c < 32; ++c) x[c] = 0.f;
      }
      v16bf HA, HB, LA, LB;
#pragma unroll
      for (int j = 0; j < 8; ++j) {
        float  a = x[j];          __bf16 ha = (__bf16)a;
        float  b = x[8 + j];      __bf16 hb = (__bf16)b;
        float  c = x[16 + j];     __bf16 hc = (__bf16)c;
        float  d = x[24 + j];     __bf16 hd = (__bf16)d;
        HA[j]     = ha;  LA[j]     = (__bf16)(a - (float)ha);
        HA[j + 8] = hb;  LA[j + 8] = (__bf16)(b - (float)hb);
        HB[j]     = hc;  LB[j]     = (__bf16)(c - (float)hc);
        HB[j + 8] = hd;  LB[j + 8] = (__bf16)(d - (float)hd);
      }
      lfe[k][chunk][msub][0][r]      = HA;
      lfe[k][chunk][msub][0][r + 16] = HB;
      lfe[k][chunk][msub][1][r]      = LA;
      lfe[k][chunk][msub][1][r + 16] = LB;
    }
  }
  __syncthreads();

  const int wave = t >> 5;
  const int lane = t & 31;
  const int nt0  = (wave & 7) * 2;
  const int ms0  = (wave >> 3) * 2;

  v8f acc00 = {};   // ms0,   nt0
  v8f acc01 = {};   // ms0,   nt0+1
  v8f acc10 = {};   // ms0+1, nt0
  v8f acc11 = {};   // ms0+1, nt0+1
  if (wave < 8) {
    compute_tile<0>(lfe, Wsw, lane, nt0, acc00, acc01, acc10, acc11);
  } else {
    compute_tile<2>(lfe, Wsw, lane, nt0, acc00, acc01, acc10, acc11);
  }

  // ---- store: C/D layout  VGPR j -> M = j + 8*(lane>>4), N = lane&15 ----
  const int nl    = lane & 15;
  const int mb    = (lane >> 4) * 8;
  const int ncol0 = nt0 * 16 + nl;
#pragma unroll
  for (int j = 0; j < 8; ++j) {
    int row0 = m0 + ms0 * 16 + mb + j;
    int row1 = row0 + 16;
    if (row0 < MOUT) {
      long r0 = (long)row0 * COUT;
      out[r0 + ncol0]      = acc00[j];
      out[r0 + ncol0 + 16] = acc01[j];
    }
    if (row1 < MOUT) {
      long r1 = (long)row1 * COUT;
      out[r1 + ncol0]      = acc10[j];
      out[r1 + ncol0 + 16] = acc11[j];
    }
  }
}

// ---------------------------------------------------------------------------
// Kernel 3: deterministic per-column partial sums (fixed order, coalesced).
// ---------------------------------------------------------------------------
#define CHUNKR 391  // ceil(100000/256)
__global__ __launch_bounds__(256) void col_reduce(const float* __restrict__ out,
                                                  float* __restrict__ pS,
                                                  float* __restrict__ pQ) {
  int b  = blockIdx.x;                         // 0..255 row chunk
  int c  = threadIdx.x;                        // column
  int r0 = b * CHUNKR;
  int r1 = r0 + CHUNKR;
  if (r1 > MOUT) r1 = MOUT;
  float s = 0.f, q = 0.f;
  for (int r = r0; r < r1; ++r) {
    float v = out[(long)r * COUT + c];
    s += v;
    q += v * v;
  }
  pS[b * COUT + c] = s;
  pQ[b * COUT + c] = q;
}

// ---------------------------------------------------------------------------
// Kernel 4: finalize BN stats -> per-column scale/shift (fixed order).
// ---------------------------------------------------------------------------
__global__ __launch_bounds__(256) void bn_finalize(
    const float* __restrict__ pS, const float* __restrict__ pQ,
    const float* __restrict__ gamma, const float* __restrict__ beta,
    float* __restrict__ scale, float* __restrict__ shift) {
  int c = threadIdx.x;
  float s = 0.f, q = 0.f;
  for (int i = 0; i < 256; ++i) {
    s += pS[i * COUT + c];
    q += pQ[i * COUT + c];
  }
  float mu  = s * (1.0f / MOUT);
  float var = q * (1.0f / MOUT) - mu * mu;
  float sc  = gamma[c] * rsqrtf(var + EPS);
  scale[c]  = sc;
  shift[c]  = beta[c] - mu * sc;
}

// ---------------------------------------------------------------------------
// Kernel 5: apply BN + ReLU in place (float4).
// ---------------------------------------------------------------------------
__global__ __launch_bounds__(256) void bn_apply(float* __restrict__ out,
                                                const float* __restrict__ scale,
                                                const float* __restrict__ shift) {
  long i = (long)blockIdx.x * 256 + threadIdx.x;   // MOUT*COUT/4 elements
  v4f v  = ((const v4f*)out)[i];
  int cq = (int)(i & (COUT / 4 - 1));
  v4f sc = ((const v4f*)scale)[cq];
  v4f sh = ((const v4f*)shift)[cq];
  v.x = fmaxf(fmaf(v.x, sc.x, sh.x), 0.f);
  v.y = fmaxf(fmaf(v.y, sc.y, sh.y), 0.f);
  v.z = fmaxf(fmaf(v.z, sc.z, sh.z), 0.f);
  v.w = fmaxf(fmaf(v.w, sc.w, sh.w), 0.f);
  ((v4f*)out)[i] = v;
}

// ---------------------------------------------------------------------------
extern "C" void kernel_launch(void* const* d_in, const int* in_sizes, int n_in,
                              void* d_out, int out_size, void* d_ws, size_t ws_size,
                              hipStream_t stream) {
  const float* feats = (const float*)d_in[0];
  const float* W     = (const float*)d_in[1];
  const float* gamma = (const float*)d_in[2];
  const float* beta  = (const float*)d_in[3];
  const int*   gidx  = (const int*)d_in[4];
  float* out = (float*)d_out;

  char* ws      = (char*)d_ws;
  v16bf* Wsw    = (v16bf*)ws;                                // 1 MB
  float* pS     = (float*)(ws + (1 << 20));                  // 256 KB
  float* pQ     = (float*)(ws + (1 << 20) + (256 << 10));    // 256 KB
  float* scale  = (float*)(ws + (1 << 20) + (512 << 10));    // 1 KB
  float* shift  = scale + COUT;                              // 1 KB

  build_wsw  <<<32768 / 256, 256, 0, stream>>>(W, Wsw);
  gemm_kernel<<<NWG, 512, 0, stream>>>(feats, gidx, Wsw, out);
  col_reduce <<<256, 256, 0, stream>>>(out, pS, pQ);
  bn_finalize<<<1, COUT, 0, stream>>>(pS, pQ, gamma, beta, scale, shift);
  bn_apply   <<<(MOUT * (COUT / 4)) / 256, 256, 0, stream>>>(out, scale, shift);
}
